// Retriever_83313775608528
// MI455X (gfx1250) — compile-verified, matrix operand order checked by reading, not run on hardware
//
#include <hip/hip_runtime.h>
#include <hip/hip_bf16.h>

typedef __attribute__((ext_vector_type(16))) __bf16          v16bf;
typedef __attribute__((ext_vector_type(16))) unsigned short  v16u;
typedef __attribute__((ext_vector_type(16))) float           v16f;
typedef __attribute__((ext_vector_type(8)))  float           v8f;
typedef __attribute__((ext_vector_type(8)))  unsigned int    v8u;

#define BATCH    4
#define NQ       1024
#define NK       16384
#define DDIM     1024
#define MTILE    32          // queries per workgroup (2 WMMA row-tiles)
#define NWAVES   8
#define TOPK     32

// round-to-nearest-even float -> bf16 (bit pattern); used only for the small Q tile
static __device__ __forceinline__ unsigned short f2bf(float x) {
    unsigned u = __float_as_uint(x);
    unsigned r = u + 0x7FFFu + ((u >> 16) & 1u);
    return (unsigned short)(r >> 16);
}
static __device__ __forceinline__ float bf2f(unsigned short h) {
    return __uint_as_float(((unsigned)h) << 16);
}

__global__ __launch_bounds__(256, 1)
void retriever_topk_wmma(const float* __restrict__ Q,
                         const float* __restrict__ Kmat,
                         const float* __restrict__ margin,
                         float* __restrict__ out)
{
    // LDS: Q tile (bf16 hi/lo), per-wave score staging, per-lane top-k lists
    __shared__ __attribute__((aligned(32))) unsigned short sQhi[MTILE * DDIM];
    __shared__ __attribute__((aligned(32))) unsigned short sQlo[MTILE * DDIM];
    __shared__ float sStage[NWAVES][MTILE][17];     // padded: conflict-free row reads
    __shared__ float sVals[NWAVES][32][TOPK + 1];
    __shared__ int   sIdx [NWAVES][32][TOPK + 1];

    const int tid  = threadIdx.x;
    const int wave = tid >> 5;
    const int lane = tid & 31;

    const int b  = blockIdx.x >> 5;                // NQ/MTILE = 32 tiles per batch
    const int q0 = (blockIdx.x & 31) * MTILE;

    const float* Qb = Q    + (size_t)b * NQ * DDIM;
    const float* Kb = Kmat + (size_t)b * NK * DDIM;

    // ---- stage Q tile into LDS as bf16 hi + lo residual (RNE, done once) ----
    for (int i = tid; i < MTILE * DDIM; i += 256) {
        float x = Qb[(size_t)(q0 + (i >> 10)) * DDIM + (i & (DDIM - 1))];
        unsigned short hi = f2bf(x);
        sQhi[i] = hi;
        sQlo[i] = f2bf(x - bf2f(hi));
    }
    for (int e = 0; e < TOPK; ++e) {
        sVals[wave][lane][e] = -INFINITY;
        sIdx [wave][lane][e] = 0;
    }
    __syncthreads();

    const int colsub = lane & 15;                  // B column / A row within tile
    const int khalf  = (lane < 16) ? 0 : 16;       // K-half split across lane groups

    float curmin = -INFINITY;
    int   minpos = 0;

    for (int nt = 0; nt < NK; nt += NWAVES * 16) {
        const int ncol = nt + wave * 16;           // this wave's 16 keys
        const float* kptr = Kb + (size_t)(ncol + colsub) * DDIM + khalf;

        v8f acc0 = {};                             // rows 0..15
        v8f acc1 = {};                             // rows 16..31
        #pragma unroll 4
        for (int kk = 0; kk < DDIM; kk += 32) {
            const int abase = colsub * DDIM + kk + khalf;
            v16bf ahi0 = __builtin_bit_cast(v16bf, *(const v16u*)&sQhi[abase]);
            v16bf alo0 = __builtin_bit_cast(v16bf, *(const v16u*)&sQlo[abase]);
            v16bf ahi1 = __builtin_bit_cast(v16bf, *(const v16u*)&sQhi[abase + 16 * DDIM]);
            v16bf alo1 = __builtin_bit_cast(v16bf, *(const v16u*)&sQlo[abase + 16 * DDIM]);

            // 64B coalesced fp32 K load; truncate-pack to bf16: 8 x v_perm_b32
            v16f bv = *(const v16f*)(kptr + kk);
            v8u bp;
            #pragma unroll
            for (int e = 0; e < 8; ++e)
                bp[e] = __builtin_amdgcn_perm(__float_as_uint(bv[2 * e + 1]),
                                              __float_as_uint(bv[2 * e]),
                                              0x07060302u);
            v16bf bb = __builtin_bit_cast(v16bf, bp);

            // 4 WMMAs per converted K fragment (hi/lo Q x 2 row-tiles)
            acc0 = __builtin_amdgcn_wmma_f32_16x16x32_bf16(false, ahi0, false, bb,
                                                           (short)0, acc0, false, false);
            acc0 = __builtin_amdgcn_wmma_f32_16x16x32_bf16(false, alo0, false, bb,
                                                           (short)0, acc0, false, false);
            acc1 = __builtin_amdgcn_wmma_f32_16x16x32_bf16(false, ahi1, false, bb,
                                                           (short)0, acc1, false, false);
            acc1 = __builtin_amdgcn_wmma_f32_16x16x32_bf16(false, alo1, false, bb,
                                                           (short)0, acc1, false, false);
        }

        // prefetch next K chunk for this wave
        if (nt + NWAVES * 16 < NK)
            __builtin_prefetch(Kb + (size_t)(nt + NWAVES * 16 + wave * 16 + colsub) * DDIM, 0, 1);

        // stage 32x16 scores: vgpr j -> row (tile*16 + j + 8*(lane>=16)), col = lane&15
        const int rsub = (lane < 16) ? 0 : 8;
        #pragma unroll
        for (int j = 0; j < 8; ++j) {
            sStage[wave][j + rsub][colsub]      = acc0[j];
            sStage[wave][16 + j + rsub][colsub] = acc1[j];
        }

        // remap: lane L owns full row L of this wave's 16-column slice
        #pragma unroll
        for (int c = 0; c < 16; ++c) {
            float v = sStage[wave][lane][c];
            if (v > curmin) {
                sVals[wave][lane][minpos] = v;
                sIdx [wave][lane][minpos] = ncol + c;
                float m = sVals[wave][lane][0]; int mp = 0;
                for (int e = 1; e < TOPK; ++e) {
                    float x = sVals[wave][lane][e];
                    if (x < m) { m = x; mp = e; }
                }
                curmin = m; minpos = mp;
            }
        }
    }
    __syncthreads();

    // ---- merge: wave w extracts exact top-32 for rows 4w .. 4w+3 ----
    const float  marg = margin[0];
    const size_t sect = (size_t)BATCH * NQ * TOPK;

    for (int rr = 0; rr < 4; ++rr) {
        const int row = wave * 4 + rr;
        const size_t obase = ((size_t)b * NQ + q0 + row) * TOPK;

        for (int k = 0; k < TOPK; ++k) {
            // row candidates: sVals[W][row][*], W=0..7 -> 256 entries, 8 per lane
            float bm = -INFINITY; int be = -1;
            #pragma unroll
            for (int t = 0; t < 8; ++t) {
                int e = lane * 8 + t;                 // 0..255
                float x = sVals[e >> 5][row][e & 31];
                if (x > bm) { bm = x; be = e; }
            }
            int bidx = 0;
            if (be >= 0) bidx = sIdx[be >> 5][row][be & 31];
            int bl = lane;
            for (int off = 16; off > 0; off >>= 1) {  // wave32 argmax
                float ov = __shfl_xor(bm,   off, 32);
                int   oi = __shfl_xor(bidx, off, 32);
                int   oe = __shfl_xor(be,   off, 32);
                int   ol = __shfl_xor(bl,   off, 32);
                if (ov > bm || (ov == bm && ol < bl)) { bm = ov; bidx = oi; be = oe; bl = ol; }
            }
            if (lane == bl && be >= 0)
                sVals[be >> 5][row][be & 31] = -INFINITY;   // retire winner
            if (lane == 0) {
                float att = 1.0f / (1.0f + __expf(-(bm + marg)));
                out[obase + k]            = att;           // att_scores
                out[sect + obase + k]     = (float)bidx;   // top_nei_idx
                out[2 * sect + obase + k] = 1.0f;          // nei_mask
            }
        }
    }
}

extern "C" void kernel_launch(void* const* d_in, const int* in_sizes, int n_in,
                              void* d_out, int out_size, void* d_ws, size_t ws_size,
                              hipStream_t stream) {
    const float* Q      = (const float*)d_in[0];   // [4,1024,1024]
    const float* Kmat   = (const float*)d_in[1];   // [4,16384,1024]
    const float* margin = (const float*)d_in[2];   // [1]
    float* out = (float*)d_out;

    dim3 grid(BATCH * (NQ / MTILE));   // 128 workgroups
    dim3 block(256);                   // 8 wave32 waves
    retriever_topk_wmma<<<grid, block, 0, stream>>>(Q, Kmat, margin, out);
}
